// PTRnet_53403623358973
// MI455X (gfx1250) — matrix operand order repaired and checked by reading, not compiled, not run on hardware
//
#include <hip/hip_runtime.h>
#include <hip/hip_bf16.h>
#include <math.h>

// ---------------- problem constants (from reference) ----------------
#define BB      4
#define LL      2048
#define KK      4
#define DD      256
#define EE      512
#define NN      16
#define RANKR   16
#define NLAY    4
#define NTOKC   14
#define NTISC   30
#define ROWS    (BB * LL)          // 8192
#define DBLW    (RANKR + 2 * NN)   // 48
#define NHEADS  19                 // 4+5+3+7
#define CH      64                 // scan chunks
#define CLEN    (LL / CH)          // 32 steps per chunk

typedef __attribute__((ext_vector_type(16))) _Float16 v16h;
typedef __attribute__((ext_vector_type(8)))  _Float16 v8h;
typedef __attribute__((ext_vector_type(8)))  float    v8f;
typedef _Float16 half_t;

union frag16 { v16h v; v8h h[2]; };

__device__ __forceinline__ float silu_f(float v) { return v / (1.f + __expf(-v)); }

// ---------------- renorm scales for embedding tables ----------------
__global__ void ptr_renorm_scales(const float* __restrict__ te, const float* __restrict__ se,
                                  float* __restrict__ tsc, float* __restrict__ ssc) {
  int i = threadIdx.x;
  if (i < NTISC) {
    float s = 0.f;
    for (int d = 0; d < DD; ++d) { float v = te[i * DD + d]; s += v * v; }
    float n = sqrtf(s);
    tsc[i] = fminf(1.f, 1.f / fmaxf(n, 1e-7f));
  } else if (i < NTISC + NTOKC) {
    int j = i - NTISC;
    float s = 0.f;
    for (int d = 0; d < DD; ++d) { float v = se[j * DD + d]; s += v * v; }
    float n = sqrtf(s);
    ssc[j] = fminf(1.f, 1.f / fmaxf(n, 1e-7f));
  }
}

// ---------------- embedding ----------------
__global__ void ptr_embed(const int* __restrict__ rna, const int* __restrict__ tis,
                          const float* __restrict__ te, const float* __restrict__ se,
                          const float* __restrict__ tsc, const float* __restrict__ ssc,
                          float* __restrict__ x32, half_t* __restrict__ x16) {
  int idx = blockIdx.x * blockDim.x + threadIdx.x;
  if (idx >= ROWS * DD) return;
  int d = idx % DD;
  int row = idx / DD;
  int b = row / LL;
  float s = 0.f;
  #pragma unroll
  for (int k = 0; k < KK; ++k) {
    int t = rna[row * KK + k];
    s += se[t * DD + d] * ssc[t];
  }
  float val = 0.f;
  if (s != 0.f) {
    int ti = tis[b];
    val = s + te[ti * DD + d] * tsc[ti];
  }
  x32[idx] = val;
  x16[idx] = (half_t)val;
}

// ---------------- pack f32 weight [K,N] into f16 WMMA-B fragment order ----------------
// Layout: [nTile][kTile][lane][16 halves]; lane = ((k%32)/16)*16 + n%16, elem j = k%16.
__global__ void ptr_pack_b(const float* __restrict__ W, half_t* __restrict__ out,
                           int K, int N) {
  int idx = blockIdx.x * blockDim.x + threadIdx.x;
  if (idx >= K * N) return;
  int n = idx % N;
  int k = idx / N;
  int nt = n >> 4, nl = n & 15;
  int kt = k >> 5, kr = k & 31;
  int lane = (kr >> 4) * 16 + nl;
  int j = kr & 15;
  size_t o = ((((size_t)nt * (K >> 5) + kt) * 32) + lane) * 16 + j;
  out[o] = (half_t)W[idx];
}

// ---------------- WMMA GEMM: C(f32)[M,N] = A(f16 row-major)[M,K] @ Bpacked ----------------
// One wave computes a 16x32 output strip (two 16x16 N-tiles sharing the A fragment).
__global__ void ptr_gemm_wmma(const half_t* __restrict__ A, const half_t* __restrict__ Bp,
                              float* __restrict__ C, half_t* __restrict__ Ch,
                              int M, int N, int K) {
  int lane = threadIdx.x & 31;
  int wave = blockIdx.x * (blockDim.x >> 5) + (threadIdx.x >> 5);
  int tilesN = N >> 4;
  int tilesM = M >> 4;
  int pairsN = (tilesN + 1) >> 1;
  if (wave >= tilesM * pairsN) return;            // wave-uniform
  int tileM = (wave / pairsN) << 4;
  int tN0   = (wave % pairsN) * 2;
  bool two  = (tN0 + 1) < tilesN;                  // wave-uniform
  int kTiles = K >> 5;

  const int mA    = tileM + (lane & 15);
  const int khalf = (lane >> 4) * 16;
  const half_t* Arow = A + (size_t)mA * K + khalf;
  const half_t* B0   = Bp + ((size_t)tN0 * kTiles * 32 + lane) * 16;
  const half_t* B1   = Bp + ((size_t)(tN0 + 1) * kTiles * 32 + lane) * 16;

  v8f acc0 = {}, acc1 = {};
  #pragma unroll 2
  for (int kt = 0; kt < kTiles; ++kt) {
    frag16 a, b0;
    a.h[0] = *(const v8h*)(Arow + kt * 32);
    a.h[1] = *(const v8h*)(Arow + kt * 32 + 8);
    b0.h[0] = *(const v8h*)(B0 + (size_t)kt * 32 * 16);
    b0.h[1] = *(const v8h*)(B0 + (size_t)kt * 32 * 16 + 8);
    acc0 = __builtin_amdgcn_wmma_f32_16x16x32_f16(false, a.v, false, b0.v, (short)0,
                                                  acc0, false, false);
    if (two) {
      frag16 b1;
      b1.h[0] = *(const v8h*)(B1 + (size_t)kt * 32 * 16);
      b1.h[1] = *(const v8h*)(B1 + (size_t)kt * 32 * 16 + 8);
      acc1 = __builtin_amdgcn_wmma_f32_16x16x32_f16(false, a.v, false, b1.v, (short)0,
                                                    acc1, false, false);
    }
  }

  int nl = lane & 15;
  int mo = (lane >> 4) * 8;
  #pragma unroll
  for (int r = 0; r < 8; ++r) {
    int row = tileM + mo + r;
    int n0 = (tN0 << 4) + nl;
    C[(size_t)row * N + n0] = acc0[r];
    if (Ch) Ch[(size_t)row * N + n0] = (half_t)acc0[r];
    if (two) {
      int n1 = n0 + 16;
      C[(size_t)row * N + n1] = acc1[r];
      if (Ch) Ch[(size_t)row * N + n1] = (half_t)acc1[r];
    }
  }
}

// ---------------- depthwise causal conv (DCONV=4) + bias + SiLU ----------------
__global__ void ptr_conv_silu(const float* __restrict__ xz, const float* __restrict__ cw,
                              const float* __restrict__ cb, float* __restrict__ xc32,
                              half_t* __restrict__ xc16) {
  int idx = blockIdx.x * blockDim.x + threadIdx.x;
  if (idx >= ROWS * EE) return;
  int e   = idx % EE;
  int row = idx / EE;
  int l   = row % LL;
  int b   = row / LL;
  float s = cb[e];
  #pragma unroll
  for (int j = 0; j < 4; ++j) {
    int ll = l - 3 + j;
    if (ll >= 0) s += xz[(size_t)(b * LL + ll) * (2 * EE) + e] * cw[e * 4 + j];
  }
  float v = silu_f(s);
  xc32[idx] = v;
  xc16[idx] = (half_t)v;
}

// ---------------- delta = softplus(dt @ W_dt + b_dt) ----------------
__global__ void ptr_delta(const float* __restrict__ dbl, const float* __restrict__ Wdt,
                          const float* __restrict__ bdt, float* __restrict__ dlt) {
  int idx = blockIdx.x * blockDim.x + threadIdx.x;
  if (idx >= ROWS * EE) return;
  int e   = idx % EE;
  int row = idx / EE;
  float s = bdt[e];
  #pragma unroll
  for (int r = 0; r < RANKR; ++r) s += dbl[row * DBLW + r] * Wdt[r * EE + e];
  dlt[idx] = (s > 20.f) ? s : log1pf(__expf(s));
}

// ---------------- chunked parallel scan ----------------
// Recurrence h = dA*h + dBx is elementwise-affine over (b,e,n); chunk over L.
// Phase 1: per (b,chunk,e): chunk-local h_part (h_in=0) and prod(dA).
__global__ void ptr_scan_part(const float* __restrict__ dbl, const float* __restrict__ dlt,
                              const float* __restrict__ xc32, const float* __restrict__ Alog,
                              float* __restrict__ prodA, float* __restrict__ hpart) {
  int b = blockIdx.x;
  int c = blockIdx.y;
  int e = blockIdx.z * blockDim.x + threadIdx.x;
  __shared__ float sB[NN];
  float A[NN], h[NN], pA[NN];
  #pragma unroll
  for (int n = 0; n < NN; ++n) {
    A[n] = -__expf(Alog[e * NN + n]);
    h[n] = 0.f;
    pA[n] = 1.f;
  }
  for (int t = 0; t < CLEN; ++t) {
    int row = b * LL + c * CLEN + t;
    if (threadIdx.x < NN) sB[threadIdx.x] = dbl[row * DBLW + RANKR + threadIdx.x];
    __syncthreads();
    float dv = dlt[(size_t)row * EE + e];
    float dx = dv * xc32[(size_t)row * EE + e];
    #pragma unroll
    for (int n = 0; n < NN; ++n) {
      float dA = __expf(dv * A[n]);
      pA[n] *= dA;
      h[n] = dA * h[n] + dx * sB[n];
    }
    __syncthreads();
  }
  size_t o = ((size_t)(b * CH + c) * EE + e) * NN;
  #pragma unroll
  for (int n = 0; n < NN; ++n) {
    prodA[o + n] = pA[n];
    hpart[o + n] = h[n];
  }
}

// Phase 2: sequential combine over chunks per (b,e,n). Rewrites prodA in place
// with each chunk's incoming state h_in (thread-private slice: no hazards).
__global__ void ptr_scan_combine(float* __restrict__ prodA, const float* __restrict__ hpart) {
  int idx = blockIdx.x * blockDim.x + threadIdx.x;   // over BB*EE*NN
  if (idx >= BB * EE * NN) return;
  int n = idx % NN;
  int e = (idx / NN) % EE;
  int b = idx / (NN * EE);
  float st = 0.f;
  for (int c = 0; c < CH; ++c) {
    size_t o = ((size_t)(b * CH + c) * EE + e) * NN + n;
    float pA = prodA[o];
    float hp = hpart[o];
    prodA[o] = st;                 // h_in for chunk c
    st = pA * st + hp;
  }
}

// Phase 3: replay each chunk from its true h_in; emit gated y (f16).
__global__ void ptr_scan_final(const float* __restrict__ dbl, const float* __restrict__ dlt,
                               const float* __restrict__ xc32, const float* __restrict__ xz,
                               const float* __restrict__ Alog, const float* __restrict__ Dp,
                               const float* __restrict__ hin, half_t* __restrict__ y16) {
  int b = blockIdx.x;
  int c = blockIdx.y;
  int e = blockIdx.z * blockDim.x + threadIdx.x;
  __shared__ float sB[NN];
  __shared__ float sC[NN];
  float A[NN], h[NN];
  size_t o = ((size_t)(b * CH + c) * EE + e) * NN;
  #pragma unroll
  for (int n = 0; n < NN; ++n) {
    A[n] = -__expf(Alog[e * NN + n]);
    h[n] = hin[o + n];
  }
  float dp = Dp[e];
  for (int t = 0; t < CLEN; ++t) {
    int row = b * LL + c * CLEN + t;
    if (threadIdx.x < NN)
      sB[threadIdx.x] = dbl[row * DBLW + RANKR + threadIdx.x];
    else if (threadIdx.x < 2 * NN)
      sC[threadIdx.x - NN] = dbl[row * DBLW + RANKR + NN + (threadIdx.x - NN)];
    __syncthreads();
    float dv = dlt[(size_t)row * EE + e];
    float xv = xc32[(size_t)row * EE + e];
    float dx = dv * xv;
    float y  = 0.f;
    #pragma unroll
    for (int n = 0; n < NN; ++n) {
      float dA = __expf(dv * A[n]);
      h[n] = dA * h[n] + dx * sB[n];
      y += h[n] * sC[n];
    }
    y += xv * dp;
    float z = xz[(size_t)row * (2 * EE) + EE + e];
    y *= silu_f(z);
    y16[(size_t)row * EE + e] = (half_t)y;
    __syncthreads();
  }
}

// ---------------- heads: logits + per-head softmax, concat to 19 cols ----------------
__global__ void ptr_heads(const float* __restrict__ x,
                          const float* W0, const float* b0, const float* W1, const float* b1,
                          const float* W2, const float* b2, const float* W3, const float* b3,
                          float* __restrict__ out) {
  int row = blockIdx.x * blockDim.x + threadIdx.x;
  if (row >= ROWS) return;
  const float* xr = x + (size_t)row * DD;
  const float* Ws[4] = {W0, W1, W2, W3};
  const float* bs[4] = {b0, b1, b2, b3};
  const int    cs[4] = {4, 5, 3, 7};
  float logits[NHEADS];
  int off = 0;
  for (int hh = 0; hh < 4; ++hh) {
    const float* W = Ws[hh];
    const float* bb = bs[hh];
    int c = cs[hh];
    float mx = -1e30f;
    for (int j = 0; j < c; ++j) {
      float s = bb[j];
      for (int d = 0; d < DD; ++d) s += xr[d] * W[d * c + j];
      logits[off + j] = s;
      mx = fmaxf(mx, s);
    }
    float den = 0.f;
    for (int j = 0; j < c; ++j) {
      float ev = __expf(logits[off + j] - mx);
      logits[off + j] = ev;
      den += ev;
    }
    float inv = 1.f / den;
    for (int j = 0; j < c; ++j) out[(size_t)row * NHEADS + off + j] = logits[off + j] * inv;
    off += c;
  }
}

// ---------------- host ----------------
extern "C" void kernel_launch(void* const* d_in, const int* in_sizes, int n_in,
                              void* d_out, int out_size, void* d_ws, size_t ws_size,
                              hipStream_t stream) {
  (void)in_sizes; (void)n_in; (void)out_size; (void)ws_size;
  // Top-level dict insertion order: rna_data_pad, tissue_id, seq_lengths, params.
  // params flattened as a JAX pytree (sorted dict keys):
  //  per layer (9 leaves): A_log, Dp, W_dt, W_in, W_out, W_x, b_dt, conv_b, conv_w
  //  layers[0..3] then preds[0..3] (W, b each) then seq_emb then tissue_emb.
  const int* rna = (const int*)d_in[0];
  const int* tis = (const int*)d_in[1];
  const int PBASE = 3;
  auto L_Alog = [&](int l) { return (const float*)d_in[PBASE + 9 * l + 0]; };
  auto L_Dp   = [&](int l) { return (const float*)d_in[PBASE + 9 * l + 1]; };
  auto L_Wdt  = [&](int l) { return (const float*)d_in[PBASE + 9 * l + 2]; };
  auto L_Win  = [&](int l) { return (const float*)d_in[PBASE + 9 * l + 3]; };
  auto L_Wout = [&](int l) { return (const float*)d_in[PBASE + 9 * l + 4]; };
  auto L_Wx   = [&](int l) { return (const float*)d_in[PBASE + 9 * l + 5]; };
  auto L_bdt  = [&](int l) { return (const float*)d_in[PBASE + 9 * l + 6]; };
  auto L_cb   = [&](int l) { return (const float*)d_in[PBASE + 9 * l + 7]; };
  auto L_cw   = [&](int l) { return (const float*)d_in[PBASE + 9 * l + 8]; };
  const int HBASE = PBASE + 9 * NLAY;
  auto H_W = [&](int h) { return (const float*)d_in[HBASE + 2 * h + 0]; };
  auto H_b = [&](int h) { return (const float*)d_in[HBASE + 2 * h + 1]; };
  const float* seq_emb = (const float*)d_in[HBASE + 8];
  const float* tis_emb = (const float*)d_in[HBASE + 9];

  size_t off = 0;
  auto carve = [&](size_t bytes) {
    char* p = (char*)d_ws + off;
    off += (bytes + 255) & ~(size_t)255;
    return (void*)p;
  };
  float*  x32   = (float*)carve((size_t)ROWS * DD * 4);
  half_t* x16   = (half_t*)carve((size_t)ROWS * DD * 2);
  float*  xz    = (float*)carve((size_t)ROWS * 2 * EE * 4);
  float*  xc32  = (float*)carve((size_t)ROWS * EE * 4);
  half_t* xc16  = (half_t*)carve((size_t)ROWS * EE * 2);
  float*  dbl   = (float*)carve((size_t)ROWS * DBLW * 4);
  float*  dlt   = (float*)carve((size_t)ROWS * EE * 4);
  half_t* y16   = (half_t*)carve((size_t)ROWS * EE * 2);
  half_t* win16 = (half_t*)carve((size_t)DD * 2 * EE * 2);   // packed B for W_in
  half_t* wx16  = (half_t*)carve((size_t)EE * DBLW * 2);     // packed B for W_x
  half_t* wo16  = (half_t*)carve((size_t)EE * DD * 2);       // packed B for W_out
  float*  prodA = (float*)carve((size_t)BB * CH * EE * NN * 4);  // then reused as h_in
  float*  hpart = (float*)carve((size_t)BB * CH * EE * NN * 4);
  float*  tsc   = (float*)carve(NTISC * 4);
  float*  ssc   = (float*)carve(NTOKC * 4);

  auto gemm = [&](const half_t* A, const half_t* Bp, float* C, half_t* Ch,
                  int M, int N, int K) {
    int pairsN = ((N >> 4) + 1) >> 1;
    int waves  = (M >> 4) * pairsN;
    int blocks = (waves * 32 + 255) / 256;
    ptr_gemm_wmma<<<blocks, 256, 0, stream>>>(A, Bp, C, Ch, M, N, K);
  };

  ptr_renorm_scales<<<1, 64, 0, stream>>>(tis_emb, seq_emb, tsc, ssc);
  ptr_embed<<<(ROWS * DD + 255) / 256, 256, 0, stream>>>(rna, tis, tis_emb, seq_emb,
                                                         tsc, ssc, x32, x16);

  for (int l = 0; l < NLAY; ++l) {
    ptr_pack_b<<<(DD * 2 * EE + 255) / 256, 256, 0, stream>>>(L_Win(l),  win16, DD, 2 * EE);
    ptr_pack_b<<<(EE * DBLW + 255) / 256, 256, 0, stream>>>(L_Wx(l),   wx16,  EE, DBLW);
    ptr_pack_b<<<(EE * DD + 255) / 256, 256, 0, stream>>>(L_Wout(l), wo16,  EE, DD);

    gemm(x16, win16, xz, nullptr, ROWS, 2 * EE, DD);                      // x @ W_in
    ptr_conv_silu<<<(ROWS * EE + 255) / 256, 256, 0, stream>>>(xz, L_cw(l), L_cb(l),
                                                               xc32, xc16);
    gemm(xc16, wx16, dbl, nullptr, ROWS, DBLW, EE);                       // xc @ W_x
    ptr_delta<<<(ROWS * EE + 255) / 256, 256, 0, stream>>>(dbl, L_Wdt(l), L_bdt(l), dlt);

    dim3 sgrid(BB, CH, EE / 256);
    ptr_scan_part<<<sgrid, 256, 0, stream>>>(dbl, dlt, xc32, L_Alog(l), prodA, hpart);
    ptr_scan_combine<<<(BB * EE * NN + 255) / 256, 256, 0, stream>>>(prodA, hpart);
    ptr_scan_final<<<sgrid, 256, 0, stream>>>(dbl, dlt, xc32, xz, L_Alog(l), L_Dp(l),
                                              prodA, y16);

    gemm(y16, wo16, x32, x16, ROWS, DD, EE);                              // y @ W_out
  }

  ptr_heads<<<(ROWS + 255) / 256, 256, 0, stream>>>(x32, H_W(0), H_b(0), H_W(1), H_b(1),
                                                    H_W(2), H_b(2), H_W(3), H_b(3),
                                                    (float*)d_out);
}
